// USLonG_35553739277024
// MI455X (gfx1250) — matrix operand order repaired and checked by reading, not compiled, not run on hardware
//
#include <hip/hip_runtime.h>

// Problem constants (match reference)
#define NN    10000
#define NIN   512
#define HH1   512
#define HH2   256
#define EDGES 320000
#define KP1   31      // k+1
#define NTILES (NN / 16)   // 625 column tiles in sim
#define SWAVES 8           // waves per sim workgroup

typedef __attribute__((ext_vector_type(8)))  __bf16 v8bf;
typedef __attribute__((ext_vector_type(16))) __bf16 v16bf;
typedef __attribute__((ext_vector_type(8)))  float  v8f;

__device__ __forceinline__ unsigned short f2bf(float f) {
    unsigned int u = __float_as_uint(f);
    unsigned int r = u + 0x7FFFu + ((u >> 16) & 1u);   // round-to-nearest-even
    return (unsigned short)(r >> 16);
}

// ---- WMMA fragment loaders (CDNA5 ISA 7.12.2 layouts, wave32) ----
// A-matrix 16x32 bf16: lane -> row = lane&15; K chunks at kb and kb+16, kb = (lane>>4)*8
__device__ __forceinline__ v16bf load_fragA(const unsigned short* tile, int lda, int lane) {
    int m  = lane & 15;
    int kb = (lane >> 4) << 3;
    const v8bf* p0 = (const v8bf*)(tile + (size_t)m * lda + kb);
    const v8bf* p1 = (const v8bf*)(tile + (size_t)m * lda + kb + 16);
    v8bf lo = *p0, hi = *p1;
    return __builtin_shufflevector(lo, hi, 0,1,2,3,4,5,6,7,8,9,10,11,12,13,14,15);
}
// B-matrix 32x16 bf16 with B stored transposed (row-major [N][K]):
// lane -> col = lane&15; K chunks at kb and kb+16, kb = k0 + (lane>>4)*8
__device__ __forceinline__ v16bf load_fragB(const unsigned short* BtTile, int ldb, int k0, int lane) {
    int n  = lane & 15;
    int kb = k0 + ((lane >> 4) << 3);
    const v8bf* p0 = (const v8bf*)(BtTile + (size_t)n * ldb + kb);
    v8bf lo = p0[0], hi = p0[2];
    return __builtin_shufflevector(lo, hi, 0,1,2,3,4,5,6,7,8,9,10,11,12,13,14,15);
}

// ---------------- conversion kernels ----------------
__global__ __launch_bounds__(256) void conv_bf(const float* __restrict__ X,
                                               unsigned short* __restrict__ Y, int n) {
    int i = blockIdx.x * 256 + threadIdx.x;
    if (i < n) Y[i] = f2bf(X[i]);
}

// W [K][N] row-major f32 -> Wt [N][K] bf16
__global__ __launch_bounds__(256) void conv_transpose_bf(const float* __restrict__ W,
                                                         unsigned short* __restrict__ Wt,
                                                         int K, int N) {
    int i = blockIdx.x * 256 + threadIdx.x;
    if (i >= K * N) return;
    int k = i / N, n = i % N;
    Wt[(size_t)n * K + k] = f2bf(W[i]);
}

// ---------------- WMMA GEMM: C = A(bf16)[M,K] @ Bt(bf16)[N,K]^T + bias, opt ReLU ----------------
// one wave (32 threads) per 16x16 output tile
__global__ __launch_bounds__(32) void wmma_gemm_nt(const unsigned short* __restrict__ A,
                                                   const unsigned short* __restrict__ Bt,
                                                   const float* __restrict__ bias,
                                                   float* __restrict__ Cf,
                                                   unsigned short* __restrict__ Cbf,
                                                   int M, int N, int K, int relu) {
    int ntn  = N >> 4;
    int m0   = (blockIdx.x / ntn) << 4;
    int n0   = (blockIdx.x % ntn) << 4;
    int lane = threadIdx.x;

    v8f c = {};
    for (int k0 = 0; k0 < K; k0 += 32) {
        v16bf a = load_fragA(A + (size_t)m0 * K + k0, K, lane);
        v16bf b = load_fragB(Bt + (size_t)n0 * K, K, k0, lane);
        c = __builtin_amdgcn_wmma_f32_16x16x32_bf16(false, a, false, b, (short)0, c, false, false);
    }
    int   n     = n0 + (lane & 15);
    float bv    = bias ? bias[n] : 0.0f;
    int   mbase = m0 + ((lane >> 4) << 3);
    for (int r = 0; r < 8; ++r) {
        float v = c[r] + bv;
        if (relu) v = fmaxf(v, 0.0f);
        size_t off = (size_t)(mbase + r) * N + n;
        if (Cf)  Cf[off]  = v;
        if (Cbf) Cbf[off] = f2bf(v);
    }
}

// ---------------- edge scatter: out[row] += w * H[col], w = vals[e] * (scale? *scale : 1) ----------------
__global__ __launch_bounds__(HH2) void spmm_scatter(const int* __restrict__ rows,
                                                    const int* __restrict__ cols,
                                                    const float* __restrict__ vals,
                                                    const float* __restrict__ H,
                                                    float* __restrict__ out,
                                                    const float* __restrict__ scale) {
    int e = blockIdx.x;
    int r = rows[e], c = cols[e];
    float w = vals[e];
    if (scale) w *= scale[0];
    int tid = threadIdx.x;
    atomicAdd(&out[(size_t)r * HH2 + tid], w * H[(size_t)c * HH2 + tid]);
}

// ---------------- row L2-normalize h_a -> Xn (bf16) ----------------
__global__ __launch_bounds__(HH2) void rownorm_bf(const float* __restrict__ H,
                                                  unsigned short* __restrict__ Xn) {
    __shared__ float red[HH2];
    int row = blockIdx.x, tid = threadIdx.x;
    float v = H[(size_t)row * HH2 + tid];
    red[tid] = v * v;
    __syncthreads();
    for (int s = HH2 / 2; s > 0; s >>= 1) {
        if (tid < s) red[tid] += red[tid + s];
        __syncthreads();
    }
    float inv = 1.0f / fmaxf(sqrtf(red[0]), 1e-12f);
    Xn[(size_t)row * HH2 + tid] = f2bf(v * inv);
}

// ---------------- fused sim = Xn @ Xn^T + streaming top-31 per row ----------------
// 8 waves per 16-row block; every wave keeps the A fragments (16 rows x K=256) in
// registers and strides over 1/8 of the 625 column tiles, maintaining a private
// top-31 list per row in its own LDS slice. Lists are merged once at the end.
// Per-wave LDS ordering uses the CDNA5 split DS counter (s_wait_dscnt).
__global__ __launch_bounds__(32 * SWAVES) void sim_topk(const unsigned short* __restrict__ Xn,
                                                        float* __restrict__ tkv_g,
                                                        int* __restrict__ tki_g) {
    __shared__ float tile[SWAVES][16 * 16];
    __shared__ float tkv[SWAVES][16][KP1];
    __shared__ int   tki[SWAVES][16][KP1];

    int lane = threadIdx.x & 31;
    int wid  = threadIdx.x >> 5;
    int m0   = blockIdx.x << 4;

    v16bf afrag[8];
    #pragma unroll
    for (int s = 0; s < 8; ++s)
        afrag[s] = load_fragA(Xn + (size_t)m0 * HH2 + s * 32, HH2, lane);

    float curmin = -1e30f; int minpos = 0;
    if (lane < 16) {
        for (int j = 0; j < KP1; ++j) { tkv[wid][lane][j] = -1e30f; tki[wid][lane][j] = 0; }
    }
    __asm__ volatile("s_wait_dscnt 0x0" ::: "memory");

    for (int tt = wid; tt < NTILES; tt += SWAVES) {
        int n0 = tt << 4;
        v8f c = {};
        #pragma unroll
        for (int s = 0; s < 8; ++s) {
            v16bf b = load_fragB(Xn + (size_t)n0 * HH2, HH2, s * 32, lane);
            c = __builtin_amdgcn_wmma_f32_16x16x32_bf16(false, afrag[s], false, b,
                                                        (short)0, c, false, false);
        }
        // spill 16x16 tile to this wave's LDS slice
        // (C layout: VGPR r -> M = r + 8*(lane>=16), N = lane&15)
        {
            int n  = lane & 15;
            int mb = (lane >> 4) << 3;
            #pragma unroll
            for (int r = 0; r < 8; ++r) tile[wid][(mb + r) * 16 + n] = c[r];
        }
        __asm__ volatile("s_wait_dscnt 0x0" ::: "memory");  // wave-local spill visible
        if (lane < 16) {
            for (int j = 0; j < 16; ++j) {
                float v = tile[wid][lane * 16 + j];
                if (v > curmin) {
                    tkv[wid][lane][minpos] = v;
                    tki[wid][lane][minpos] = n0 + j;
                    float mn = tkv[wid][lane][0]; int mp = 0;
                    #pragma unroll
                    for (int q = 1; q < KP1; ++q) {
                        float x = tkv[wid][lane][q];
                        if (x < mn) { mn = x; mp = q; }
                    }
                    curmin = mn; minpos = mp;
                }
            }
        }
        __asm__ volatile("s_wait_dscnt 0x0" ::: "memory");
    }

    __syncthreads();   // convergent point: all waves finished their tile ranges

    // merge SWAVES partial lists into list 0 (threads 0..15, one per row)
    if (threadIdx.x < 16) {
        int t = threadIdx.x;
        float mn = tkv[0][t][0]; int mp = 0;
        #pragma unroll
        for (int q = 1; q < KP1; ++q) {
            float x = tkv[0][t][q];
            if (x < mn) { mn = x; mp = q; }
        }
        for (int w = 1; w < SWAVES; ++w) {
            for (int j = 0; j < KP1; ++j) {
                float v = tkv[w][t][j];
                if (v > mn) {
                    tkv[0][t][mp] = v;
                    tki[0][t][mp] = tki[w][t][j];
                    mn = tkv[0][t][0]; mp = 0;
                    #pragma unroll
                    for (int q = 1; q < KP1; ++q) {
                        float x = tkv[0][t][q];
                        if (x < mn) { mn = x; mp = q; }
                    }
                }
            }
        }
        for (int j = 0; j < KP1; ++j) {
            tkv_g[(size_t)(m0 + t) * KP1 + j] = tkv[0][t][j];
            tki_g[(size_t)(m0 + t) * KP1 + j] = tki[0][t][j];
        }
    }
}

// ---------------- degree normalization accumulation ----------------
__global__ __launch_bounds__(256) void knn_norm(const float* __restrict__ tkv,
                                                const int* __restrict__ tki,
                                                float* __restrict__ norm) {
    int row = blockIdx.x * 256 + threadIdx.x;
    if (row >= NN) return;
    float s = 0.0f;
    for (int j = 0; j < KP1; ++j) {
        float v = tkv[(size_t)row * KP1 + j];
        s += v;
        atomicAdd(&norm[tki[(size_t)row * KP1 + j]], v);
    }
    atomicAdd(&norm[row], s);   // norm = norm_row + norm_col
}

// ---------------- knn symmetric scatter: h_p += (1-t)*relu(v') * (both directions) ----------------
__global__ __launch_bounds__(HH2) void knn_scatter(const float* __restrict__ tkv,
                                                   const int* __restrict__ tki,
                                                   const float* __restrict__ norm,
                                                   const float* __restrict__ Ha,
                                                   float* __restrict__ out,
                                                   const float* __restrict__ tptr) {
    int e   = blockIdx.x;           // 0 .. NN*KP1-1
    int row = e / KP1;
    int col = tki[e];
    float v  = tkv[e];
    float vp = v * rsqrtf(norm[row]) * rsqrtf(norm[col]);
    float w  = (1.0f - tptr[0]) * fmaxf(vp, 0.0f);
    int tid = threadIdx.x;
    atomicAdd(&out[(size_t)row * HH2 + tid], w * Ha[(size_t)col * HH2 + tid]);
    atomicAdd(&out[(size_t)col * HH2 + tid], w * Ha[(size_t)row * HH2 + tid]);
}

// =====================================================================
extern "C" void kernel_launch(void* const* d_in, const int* in_sizes, int n_in,
                              void* d_out, int out_size, void* d_ws, size_t ws_size,
                              hipStream_t stream) {
    const float* feature  = (const float*)d_in[0];
    const int*   adj_rows = (const int*)  d_in[1];
    const int*   adj_cols = (const int*)  d_in[2];
    const float* adj_vals = (const float*)d_in[3];
    const float* W1       = (const float*)d_in[4];
    const float* b1       = (const float*)d_in[5];
    const float* W2       = (const float*)d_in[6];
    const float* b2       = (const float*)d_in[7];
    const float* t_ptr    = (const float*)d_in[8];
    (void)d_in[9]; (void)in_sizes; (void)n_in; (void)out_size; (void)ws_size;

    float* out  = (float*)d_out;
    float* h_s  = out;                       // [NN, HH2]
    float* h_p  = out + (size_t)NN * HH2;    // [NN, HH2]

    // workspace carve-out (256B aligned)
    char* ws = (char*)d_ws;
    size_t off = 0;
    auto carve = [&](size_t bytes) -> void* {
        off = (off + 255) & ~(size_t)255;
        void* p = ws + off;
        off += bytes;
        return p;
    };
    unsigned short* featbf = (unsigned short*)carve((size_t)NN * NIN * 2);
    unsigned short* W1t    = (unsigned short*)carve((size_t)NIN * HH1 * 2);
    unsigned short* W2t    = (unsigned short*)carve((size_t)HH1 * HH2 * 2);
    unsigned short* H1bf   = (unsigned short*)carve((size_t)NN * HH1 * 2);
    float*          h_a    = (float*)carve((size_t)NN * HH2 * 4);
    unsigned short* Xnbf   = (unsigned short*)carve((size_t)NN * HH2 * 2);
    float*          tkv    = (float*)carve((size_t)NN * KP1 * 4);
    int*            tki    = (int*)carve((size_t)NN * KP1 * 4);
    float*          norm   = (float*)carve((size_t)NN * 4);

    // 1) convert inputs to bf16 (W transposed for contiguous B-fragment loads)
    conv_bf<<<((size_t)NN * NIN + 255) / 256, 256, 0, stream>>>(feature, featbf, NN * NIN);
    conv_transpose_bf<<<((size_t)NIN * HH1 + 255) / 256, 256, 0, stream>>>(W1, W1t, NIN, HH1);
    conv_transpose_bf<<<((size_t)HH1 * HH2 + 255) / 256, 256, 0, stream>>>(W2, W2t, HH1, HH2);

    // 2) H1 = relu(feature @ W1 + b1)  -> bf16
    wmma_gemm_nt<<<(NN / 16) * (HH1 / 16), 32, 0, stream>>>(featbf, W1t, b1,
                                                            nullptr, H1bf, NN, HH1, NIN, 1);
    // 3) h_s = H1 @ W2 + b2  -> f32 (first half of d_out)
    wmma_gemm_nt<<<(NN / 16) * (HH2 / 16), 32, 0, stream>>>(H1bf, W2t, b2,
                                                            h_s, nullptr, NN, HH2, HH1, 0);
    // 4) h_a = spmm(adj, h_s)
    hipMemsetAsync(h_a, 0, (size_t)NN * HH2 * 4, stream);
    spmm_scatter<<<EDGES, HH2, 0, stream>>>(adj_rows, adj_cols, adj_vals, h_s, h_a, nullptr);

    // 5) Xn = normalize_rows(h_a) -> bf16
    rownorm_bf<<<NN, HH2, 0, stream>>>(h_a, Xnbf);

    // 6) fused sim + top-31 (dominant 51.2 GFLOP, bf16 WMMA, L2-resident, 5000 waves)
    sim_topk<<<NN / 16, 32 * SWAVES, 0, stream>>>(Xnbf, tkv, tki);

    // 7) degree normalization
    hipMemsetAsync(norm, 0, (size_t)NN * 4, stream);
    knn_norm<<<(NN + 255) / 256, 256, 0, stream>>>(tkv, tki, norm);

    // 8) h_p = (1-t)*spmm(sym knn, h_a) + t*spmm(adj, h_a)
    hipMemsetAsync(h_p, 0, (size_t)NN * HH2 * 4, stream);
    knn_scatter<<<NN * KP1, HH2, 0, stream>>>(tkv, tki, norm, h_a, h_p, t_ptr);
    spmm_scatter<<<EDGES, HH2, 0, stream>>>(adj_rows, adj_cols, adj_vals, h_a, h_p, t_ptr);
}
